// WanS2VModel_64776696758745
// MI455X (gfx1250) — matrix operand order repaired and checked by reading, not compiled, not run on hardware
//
#include <hip/hip_runtime.h>
#include <hip/hip_bf16.h>
#include <math.h>

// ---------------------------------------------------------------------------
// Types for CDNA5 WMMA (gfx1250, wave32)
// ---------------------------------------------------------------------------
typedef __bf16 bf16x8  __attribute__((ext_vector_type(8)));
typedef __bf16 bf16x16 __attribute__((ext_vector_type(16)));
typedef float  v8f     __attribute__((ext_vector_type(8)));
typedef float  f32x4   __attribute__((ext_vector_type(4)));

#define SD     2048   // sequence length S
#define DD     2048   // model dim D
#define NH     32     // heads
#define HDIM   64     // head dim
#define CTX    512
#define NFR    4      // frames
#define AUD    5      // audio tokens per frame
#define AKPAD  32     // padded audio kv
#define EPSF   1e-6f

__device__ __forceinline__ v8f wmma_bf16(bf16x16 a, bf16x16 b, v8f c) {
  return __builtin_amdgcn_wmma_f32_16x16x32_bf16(false, a, false, b, (short)0, c,
                                                 false, false);
}

__device__ __forceinline__ bf16x16 make_frag(const __bf16* lo, const __bf16* hi) {
  bf16x8 a = *(const bf16x8*)lo;
  bf16x8 b = *(const bf16x8*)hi;
  return __builtin_shufflevector(a, b, 0, 1, 2, 3, 4, 5, 6, 7,
                                 8, 9, 10, 11, 12, 13, 14, 15);
}

__device__ __forceinline__ unsigned pack_bf16(float a, float b) {
  unsigned short x = __builtin_bit_cast(unsigned short, (__bf16)a);
  unsigned short y = __builtin_bit_cast(unsigned short, (__bf16)b);
  return (unsigned)x | ((unsigned)y << 16);
}

__device__ __forceinline__ float gelu_tanh(float x) {
  float x3 = x * x * x;
  return 0.5f * x * (1.0f + tanhf(0.7978845608028654f * (x + 0.044715f * x3)));
}

// async global -> LDS copy of 16 bytes per lane (CDNA5, ASYNCcnt-tracked)
__device__ __forceinline__ void async_copy_b128(unsigned lds_addr,
                                                const void* gaddr) {
  unsigned long long ga = (unsigned long long)gaddr;
  asm volatile("global_load_async_to_lds_b128 %0, %1, off"
               :: "v"(lds_addr), "v"(ga) : "memory");
}
__device__ __forceinline__ void wait_async() {
  asm volatile("s_wait_asynccnt 0" ::: "memory");
}
__device__ __forceinline__ unsigned lds_off(const void* p) {
  return (unsigned)(size_t)p;  // low 32 bits of flat LDS pointer = LDS address
}

// ---------------------------------------------------------------------------
// RoPE tables: cos/sin (S x 32); freq split 12/10/10 over (f,h,w)
// ---------------------------------------------------------------------------
__global__ void rope_tables_kernel(float* __restrict__ cosb, float* __restrict__ sinb) {
  int idx = blockIdx.x * blockDim.x + threadIdx.x;
  if (idx >= SD * 32) return;
  int s = idx >> 5, i = idx & 31;
  int w = s & 31, h = (s >> 5) & 15, f = s >> 9;
  float pos, j, dim;
  if (i < 12)       { pos = (float)f; j = (float)i;        dim = 12.f; }
  else if (i < 22)  { pos = (float)h; j = (float)(i - 12); dim = 10.f; }
  else              { pos = (float)w; j = (float)(i - 22); dim = 10.f; }
  float ang = pos * powf(10000.f, -j / dim);
  cosb[idx] = cosf(ang);
  sinb[idx] = sinf(ang);
}

// ---------------------------------------------------------------------------
// adaLN: out = LN(x) * (1 + sc) + sh  (sc/sh from mod[l]+t_emb); isc<0 -> plain LN
// ---------------------------------------------------------------------------
__global__ __launch_bounds__(256) void adaln_kernel(
    const float* __restrict__ x, float* __restrict__ out, int M,
    const float* __restrict__ modl, const float* __restrict__ temb,
    int isc, int ish) {
  __shared__ float rs_[256], rq_[256];
  int row = blockIdx.x, tid = threadIdx.x;
  if (row >= M) return;
  const float* xr = x + (size_t)row * DD;
  float s = 0.f, q = 0.f;
  float lv[8];
#pragma unroll
  for (int i = 0; i < 8; ++i) {
    float v = xr[tid + i * 256];
    lv[i] = v; s += v; q += v * v;
  }
  rs_[tid] = s; rq_[tid] = q;
  __syncthreads();
  for (int st = 128; st > 0; st >>= 1) {
    if (tid < st) { rs_[tid] += rs_[tid + st]; rq_[tid] += rq_[tid + st]; }
    __syncthreads();
  }
  float mean = rs_[0] * (1.f / DD);
  float var  = rq_[0] * (1.f / DD) - mean * mean;
  float rstd = rsqrtf(var + EPSF);
  float* orow = out + (size_t)row * DD;
#pragma unroll
  for (int i = 0; i < 8; ++i) {
    int d = tid + i * 256;
    float nrm = (lv[i] - mean) * rstd;
    if (isc >= 0) {
      float sc = modl[isc * DD + d] + temb[isc * DD + d];
      float sh = modl[ish * DD + d] + temb[ish * DD + d];
      nrm = nrm * (1.f + sc) + sh;
    }
    orow[d] = nrm;
  }
}

// ---------------------------------------------------------------------------
// GEMM: C[M,N] = A[M,K](f32) * B[K,N](f32), bf16 WMMA, fused bias/GELU
// block = 256 thr = 8 waves; tile 128x128x32; wave tile 64x32 (4x2 WMMA frags)
// ---------------------------------------------------------------------------
__global__ __launch_bounds__(256) void gemm_bf16_kernel(
    const float* __restrict__ A, const float* __restrict__ B,
    float* __restrict__ C, int M, int N, int K,
    const float* __restrict__ bias, int act_gelu) {
  __shared__ __bf16 sA[128 * 40];
  __shared__ __bf16 sB[128 * 40];
  int tid = threadIdx.x, lane = tid & 31, w = tid >> 5;
  int wr = w >> 2, wc = w & 3;                 // 2 x 4 wave grid
  int m0 = blockIdx.y * 128, n0 = blockIdx.x * 128;

  v8f acc[4][2];
#pragma unroll
  for (int mi = 0; mi < 4; ++mi)
#pragma unroll
    for (int ni = 0; ni < 2; ++ni) acc[mi][ni] = (v8f){};

  int arow = tid >> 1, acol = (tid & 1) * 16;  // A-stage role
  int bkp  = tid >> 4, bnb  = (tid & 15) * 8;  // B-stage role (k-pair, 8 cols)
  int half = lane >> 4, l16 = lane & 15;

  for (int kt = 0; kt < K; kt += 32) {
    // stage A tile (f32 -> bf16), row-major [128 x 32], two b128 LDS stores
    {
      int gm = m0 + arow;
      float vals[16];
      if (gm < M) {
        const f32x4* ap = (const f32x4*)(A + (size_t)gm * K + kt + acol);
#pragma unroll
        for (int c = 0; c < 4; ++c) {
          f32x4 v = ap[c];
          vals[c * 4 + 0] = v.x; vals[c * 4 + 1] = v.y;
          vals[c * 4 + 2] = v.z; vals[c * 4 + 3] = v.w;
        }
        if (kt + 32 < K) __builtin_prefetch(A + (size_t)gm * K + kt + 32 + acol, 0, 1);
      } else {
#pragma unroll
        for (int c = 0; c < 16; ++c) vals[c] = 0.f;
      }
      bf16x8 v0, v1;
#pragma unroll
      for (int c = 0; c < 8; ++c) { v0[c] = (__bf16)vals[c]; v1[c] = (__bf16)vals[c + 8]; }
      __bf16* dst = sA + arow * 40 + acol;
      *(bf16x8*)dst = v0;
      *(bf16x8*)(dst + 8) = v1;
    }
    // stage B tile transposed (f32 -> bf16): sB[n][k]; pack k-pairs -> b32 stores
    {
      const float* b0 = B + (size_t)(kt + 2 * bkp) * N + n0 + bnb;
      f32x4 x0 = ((const f32x4*)b0)[0], x1 = ((const f32x4*)b0)[1];
      f32x4 y0 = ((const f32x4*)(b0 + N))[0], y1 = ((const f32x4*)(b0 + N))[1];
      if (kt + 32 < K) __builtin_prefetch(b0 + (size_t)32 * N, 0, 1);
      unsigned* sB32 = (unsigned*)sB;
      float r0[8] = {x0.x, x0.y, x0.z, x0.w, x1.x, x1.y, x1.z, x1.w};
      float r1[8] = {y0.x, y0.y, y0.z, y0.w, y1.x, y1.y, y1.z, y1.w};
#pragma unroll
      for (int c = 0; c < 8; ++c)
        sB32[(bnb + c) * 20 + bkp] = pack_bf16(r0[c], r1[c]);
    }
    __syncthreads();

    bf16x16 af[4], bfg[2];
#pragma unroll
    for (int mi = 0; mi < 4; ++mi) {
      const __bf16* p = sA + (wr * 64 + mi * 16 + l16) * 40 + half * 8;
      af[mi] = make_frag(p, p + 16);
    }
#pragma unroll
    for (int ni = 0; ni < 2; ++ni) {
      const __bf16* p = sB + (wc * 32 + ni * 16 + l16) * 40 + half * 8;
      bfg[ni] = make_frag(p, p + 16);
    }
#pragma unroll
    for (int mi = 0; mi < 4; ++mi)
#pragma unroll
      for (int ni = 0; ni < 2; ++ni)
        acc[mi][ni] = wmma_bf16(af[mi], bfg[ni], acc[mi][ni]);
    __syncthreads();
  }

  // epilogue: C layout — lane l, vgpr r: m = (l>>4)*8 + r, n = l&15
#pragma unroll
  for (int mi = 0; mi < 4; ++mi)
#pragma unroll
    for (int ni = 0; ni < 2; ++ni) {
      v8f c = acc[mi][ni];
      int gn = n0 + wc * 32 + ni * 16 + l16;
#pragma unroll
      for (int r = 0; r < 8; ++r) {
        int gm = m0 + wr * 64 + mi * 16 + half * 8 + r;
        if (gm < M) {
          float v = c[r] + (bias ? bias[gn] : 0.f);
          if (act_gelu) v = gelu_tanh(v);
          C[(size_t)gm * N + gn] = v;
        }
      }
    }
}

// ---------------------------------------------------------------------------
// RMS-norm over D + gain + RoPE, repack f32 (S,D) -> bf16 (N,S,HD)
// ---------------------------------------------------------------------------
__global__ __launch_bounds__(256) void rmsrope_kernel(
    const float* __restrict__ src, const float* __restrict__ gain,
    const float* __restrict__ cosb, const float* __restrict__ sinb,
    __bf16* __restrict__ dst) {
  __shared__ float red[256];
  int s = blockIdx.x, tid = threadIdx.x;
  const float* xr = src + (size_t)s * DD;
  float q = 0.f;
#pragma unroll
  for (int i = 0; i < 8; ++i) { float v = xr[tid + i * 256]; q += v * v; }
  red[tid] = q;
  __syncthreads();
  for (int st = 128; st > 0; st >>= 1) {
    if (tid < st) red[tid] += red[tid + st];
    __syncthreads();
  }
  float rms = rsqrtf(red[0] * (1.f / DD) + EPSF);
#pragma unroll
  for (int p = 0; p < 4; ++p) {
    int pi = tid + p * 256;            // pair index 0..1023
    int d0 = pi * 2, d1 = d0 + 1;
    float v0 = xr[d0] * rms * gain[d0];
    float v1 = xr[d1] * rms * gain[d1];
    int n = d0 >> 6, t = d0 & 63, i = (d0 >> 1) & 31;
    float c = cosb[s * 32 + i], sn = sinb[s * 32 + i];
    float o0 = v0 * c - v1 * sn;
    float o1 = v0 * sn + v1 * c;
    __bf16* dp = dst + (((size_t)n * SD + s) << 6) + t;
    dp[0] = (__bf16)o0;
    dp[1] = (__bf16)o1;
  }
}

// ---------------------------------------------------------------------------
// layout converters (f32 -> bf16)
// ---------------------------------------------------------------------------
// q: (S,D) -> (f*N + n, SQ, HD)
__global__ void convq_kernel(const float* __restrict__ src, __bf16* __restrict__ dst,
                             int SQ, int total) {
  int idx = blockIdx.x * blockDim.x + threadIdx.x;
  if (idx >= total) return;
  int s = idx >> 11, d = idx & 2047;
  int n = d >> 6, hd = d & 63;
  int f = s / SQ, sq = s - f * SQ;
  dst[(((size_t)(f * NH + n) * SQ + sq) << 6) + hd] = (__bf16)src[idx];
}
// k: src rows (f*kv_src + j, D) -> (f*N + n, SKpad, HD), zero pad j>=kv_len
__global__ void convk_kernel(const float* __restrict__ src, __bf16* __restrict__ dst,
                             int SKpad, int kv_len, int kv_src, int total) {
  int idx = blockIdx.x * blockDim.x + threadIdx.x;
  if (idx >= total) return;
  int b = idx / (SKpad * 64);
  int rem = idx - b * SKpad * 64;
  int j = rem >> 6, hd = rem & 63;
  int f = b / NH, n = b - f * NH;
  float v = (j < kv_len) ? src[(size_t)(f * kv_src + j) * DD + n * 64 + hd] : 0.f;
  dst[idx] = (__bf16)v;
}
// v: -> (f*N + n, HD, SKpad) (transposed), zero pad
__global__ void convv_kernel(const float* __restrict__ src, __bf16* __restrict__ dst,
                             int SKpad, int kv_len, int kv_src, int total) {
  int idx = blockIdx.x * blockDim.x + threadIdx.x;
  if (idx >= total) return;
  int b = idx / (SKpad * 64);
  int rem = idx - b * SKpad * 64;
  int hd = rem / SKpad, j = rem - hd * SKpad;
  int f = b / NH, n = b - f * NH;
  float v = (j < kv_len) ? src[(size_t)(f * kv_src + j) * DD + n * 64 + hd] : 0.f;
  dst[idx] = (__bf16)v;
}

// ---------------------------------------------------------------------------
// Flash attention (online softmax) with WMMA QK^T and PV.
// K/V tiles staged into LDS once per block via async global->LDS DMA
// (global_load_async_to_lds_b128 + s_wait_asynccnt), removing 8x redundant
// per-wave global streams.
// q:(BH,SQ,64) k:(BH,SKpad,64) v:(BH,64,SKpad) bf16; o:(f*SQ+sq, n*64+hd) f32
// block = 256 thr = 8 waves, wave = 16 queries; grid = (SQ/128, BH)
// ---------------------------------------------------------------------------
__global__ __launch_bounds__(256) void flash_attn_kernel(
    const __bf16* __restrict__ q, const __bf16* __restrict__ k,
    const __bf16* __restrict__ v, float* __restrict__ o,
    int SQ, int SKpad, int kv_len, float scale) {
  __shared__ __bf16 sK[32 * 64];          // (key, hd)
  __shared__ __bf16 sV[64 * 32];          // (hd, key)
  __shared__ __bf16 lds_p[8][16][40];
  int tid = threadIdx.x, lane = tid & 31, w = tid >> 5;
  int half = lane >> 4, l16 = lane & 15;
  int b = blockIdx.y;
  int f = b / NH, n = b - f * NH;
  int qrow0 = blockIdx.x * 128 + w * 16;

  // Q A-fragments (K-dim = HD = 64 -> 2 chunks of 32), loaded once from global
  bf16x16 aq[2];
  {
    const __bf16* qp = q + (((size_t)b * SQ + qrow0 + l16) << 6) + half * 8;
#pragma unroll
    for (int c = 0; c < 2; ++c)
      aq[c] = make_frag(qp + c * 32, qp + c * 32 + 16);
  }

  v8f o_acc[4];
#pragma unroll
  for (int nc = 0; nc < 4; ++nc) o_acc[nc] = (v8f){};
  float m_r[8], l_r[8];
#pragma unroll
  for (int r = 0; r < 8; ++r) { m_r[r] = -1e30f; l_r[r] = 0.f; }

  // async staging roles (16B per lane per tile)
  int kkey = tid >> 3, kcb = (tid & 7) * 8;   // K: 32 rows x 64
  int vhd  = tid >> 2, vcb = (tid & 3) * 8;   // V: 64 rows x 32

  for (int kv0 = 0; kv0 < SKpad; kv0 += 32) {
    __syncthreads();  // WAR: previous iteration done reading tiles
    async_copy_b128(lds_off(sK + kkey * 64 + kcb),
                    k + (((size_t)b * SKpad + kv0 + kkey) << 6) + kcb);
    async_copy_b128(lds_off(sV + vhd * 32 + vcb),
                    v + ((size_t)b * 64 + vhd) * SKpad + kv0 + vcb);
    wait_async();
    __syncthreads();  // tiles visible to all waves

    // QK^T -> two 16x16 score fragments (keys kv0..kv0+31)
    v8f s0 = (v8f){}, s1 = (v8f){};
#pragma unroll
    for (int nc = 0; nc < 2; ++nc) {
      const __bf16* kp = sK + (nc * 16 + l16) * 64 + half * 8;
      bf16x16 kf0 = make_frag(kp, kp + 16);
      bf16x16 kf1 = make_frag(kp + 32, kp + 48);
      if (nc == 0) { s0 = wmma_bf16(aq[0], kf0, s0); s0 = wmma_bf16(aq[1], kf1, s0); }
      else         { s1 = wmma_bf16(aq[0], kf0, s1); s1 = wmma_bf16(aq[1], kf1, s1); }
    }
    bool v0ok = (kv0 + l16) < kv_len;
    bool v1ok = (kv0 + 16 + l16) < kv_len;
    // online softmax per row (rows split across lane halves)
#pragma unroll
    for (int r = 0; r < 8; ++r) {
      float a = v0ok ? s0[r] * scale : -1e30f;
      float c = v1ok ? s1[r] * scale : -1e30f;
      float mx = fmaxf(a, c);
#pragma unroll
      for (int msk = 1; msk < 16; msk <<= 1) mx = fmaxf(mx, __shfl_xor(mx, msk, 32));
      float mnew = fmaxf(m_r[r], mx);
      float p0 = __expf(a - mnew), p1 = __expf(c - mnew);
      float rs = p0 + p1;
#pragma unroll
      for (int msk = 1; msk < 16; msk <<= 1) rs += __shfl_xor(rs, msk, 32);
      float alpha = __expf(m_r[r] - mnew);
      l_r[r] = l_r[r] * alpha + rs;
      m_r[r] = mnew;
      s0[r] = p0; s1[r] = p1;
#pragma unroll
      for (int nc = 0; nc < 4; ++nc) o_acc[nc][r] *= alpha;
    }
    // P: C-layout -> LDS -> A-fragment (wave-private region, dscnt fence)
#pragma unroll
    for (int r = 0; r < 8; ++r) {
      int ml = half * 8 + r;
      lds_p[w][ml][l16]      = (__bf16)s0[r];
      lds_p[w][ml][16 + l16] = (__bf16)s1[r];
    }
    asm volatile("s_wait_dscnt 0" ::: "memory");
    const __bf16* pb = &lds_p[w][l16][half * 8];
    bf16x16 ap = make_frag(pb, pb + 16);
    // PV: O(16x64) += P(16x32) x V(32x64)
#pragma unroll
    for (int nc = 0; nc < 4; ++nc) {
      const __bf16* vp = sV + (nc * 16 + l16) * 32 + half * 8;
      bf16x16 vf = make_frag(vp, vp + 16);
      o_acc[nc] = wmma_bf16(ap, vf, o_acc[nc]);
    }
  }

  // normalize + store
#pragma unroll
  for (int nc = 0; nc < 4; ++nc) {
    int hd = nc * 16 + l16;
#pragma unroll
    for (int r = 0; r < 8; ++r) {
      int sg = f * SQ + qrow0 + half * 8 + r;
      o[(size_t)sg * DD + n * 64 + hd] = o_acc[nc][r] / l_r[r];
    }
  }
}

// ---------------------------------------------------------------------------
// residual: x += gate * t  (gate from mod[l]+t_emb at index ig, or 1 if ig<0)
// ---------------------------------------------------------------------------
__global__ void residual_kernel(float* __restrict__ x, const float* __restrict__ t,
                                const float* __restrict__ modl,
                                const float* __restrict__ temb, int ig, int total) {
  int idx = blockIdx.x * blockDim.x + threadIdx.x;
  if (idx >= total) return;
  int d = idx & 2047;
  float g = (ig >= 0) ? (modl[ig * DD + d] + temb[ig * DD + d]) : 1.f;
  x[idx] += g * t[idx];
}

// ---------------------------------------------------------------------------
// host-side orchestration
// ---------------------------------------------------------------------------
extern "C" void kernel_launch(void* const* d_in, const int* in_sizes, int n_in,
                              void* d_out, int out_size, void* d_ws, size_t ws_size,
                              hipStream_t stream) {
  const float* in_x    = (const float*)d_in[0];
  const float* t_emb   = (const float*)d_in[1];
  const float* context = (const float*)d_in[2];
  const float* audio   = (const float*)d_in[3];
  const float* w_q  = (const float*)d_in[4];
  const float* w_k  = (const float*)d_in[5];
  const float* w_v  = (const float*)d_in[6];
  const float* w_o  = (const float*)d_in[7];
  const float* qn   = (const float*)d_in[8];
  const float* kn   = (const float*)d_in[9];
  const float* cw_q = (const float*)d_in[10];
  const float* cw_k = (const float*)d_in[11];
  const float* cw_v = (const float*)d_in[12];
  const float* cw_o = (const float*)d_in[13];
  const float* aw_q = (const float*)d_in[14];
  const float* aw_k = (const float*)d_in[15];
  const float* aw_v = (const float*)d_in[16];
  const float* aw_o = (const float*)d_in[17];
  const float* w1   = (const float*)d_in[18];
  const float* b1   = (const float*)d_in[19];
  const float* w2   = (const float*)d_in[20];
  const float* b2   = (const float*)d_in[21];
  const float* mod  = (const float*)d_in[22];

  float* x = (float*)d_out;                       // (S, D), updated in place
  const size_t SDsz = (size_t)SD * DD;

  // workspace bump allocation (float units)
  float* W = (float*)d_ws;
  size_t off = 0;
  auto alloc = [&](size_t nflt) { float* p = W + off; off += nflt; return p; };
  float* cosb = alloc((size_t)SD * 32);
  float* sinb = alloc((size_t)SD * 32);
  float* h    = alloc(SDsz);
  float* t0   = alloc(SDsz);
  float* t1   = alloc(SDsz);
  float* t2   = alloc(SDsz);      // also attention O
  float* t3   = alloc(SDsz);
  float* mid  = alloc(SDsz * 4);  // MLP hidden (S, 8192)
  float* ctk  = alloc((size_t)CTX * DD);
  float* ctv  = alloc((size_t)CTX * DD);
  float* aln  = alloc((size_t)NFR * AUD * DD);
  float* audk = alloc((size_t)NFR * AUD * DD);
  float* audv = alloc((size_t)NFR * AUD * DD);
  __bf16* qb  = (__bf16*)alloc(SDsz / 2);
  __bf16* kb  = (__bf16*)alloc(SDsz / 2);
  __bf16* vb  = (__bf16*)alloc(SDsz / 2);
  __bf16* ckb = (__bf16*)alloc((size_t)NH * CTX * HDIM / 2);
  __bf16* cvb = (__bf16*)alloc((size_t)NH * CTX * HDIM / 2);
  __bf16* akb = (__bf16*)alloc((size_t)NFR * NH * AKPAD * HDIM / 2);
  __bf16* avb = (__bf16*)alloc((size_t)NFR * NH * AKPAD * HDIM / 2);

  hipMemcpyAsync(x, in_x, SDsz * sizeof(float), hipMemcpyDeviceToDevice, stream);

  rope_tables_kernel<<<(SD * 32 + 255) / 256, 256, 0, stream>>>(cosb, sinb);

  const float scale = 0.125f;  // HD^-0.5
  auto gemm = [&](const float* A, const float* B, float* C, int M, int N, int K,
                  const float* bias, int gelu) {
    dim3 g(N / 128, (M + 127) / 128);
    gemm_bf16_kernel<<<g, 256, 0, stream>>>(A, B, C, M, N, K, bias, gelu);
  };

  for (int l = 0; l < 2; ++l) {
    const float* modl = mod + (size_t)l * 6 * DD;
    const size_t wOff = (size_t)l * DD * DD;
    const size_t mOff = (size_t)l * DD * 4 * DD;

    // ---- self attention -------------------------------------------------
    adaln_kernel<<<SD, 256, 0, stream>>>(x, h, SD, modl, t_emb, 1, 0);
    gemm(h, w_q + wOff, t0, SD, DD, DD, nullptr, 0);
    gemm(h, w_k + wOff, t1, SD, DD, DD, nullptr, 0);
    gemm(h, w_v + wOff, t2, SD, DD, DD, nullptr, 0);
    rmsrope_kernel<<<SD, 256, 0, stream>>>(t0, qn + (size_t)l * DD, cosb, sinb, qb);
    rmsrope_kernel<<<SD, 256, 0, stream>>>(t1, kn + (size_t)l * DD, cosb, sinb, kb);
    {
      int tot = NH * SD * HDIM;
      convv_kernel<<<(tot + 255) / 256, 256, 0, stream>>>(t2, vb, SD, SD, SD, tot);
    }
    flash_attn_kernel<<<dim3(SD / 128, NH), 256, 0, stream>>>(qb, kb, vb, t2,
                                                              SD, SD, SD, scale);
    gemm(t2, w_o + wOff, t3, SD, DD, DD, nullptr, 0);
    residual_kernel<<<(int)(SDsz / 256), 256, 0, stream>>>(x, t3, modl, t_emb, 2,
                                                           (int)SDsz);

    // ---- cross attention ------------------------------------------------
    adaln_kernel<<<SD, 256, 0, stream>>>(x, h, SD, modl, t_emb, -1, -1);
    gemm(h, cw_q + wOff, t0, SD, DD, DD, nullptr, 0);
    convq_kernel<<<(int)(SDsz / 256), 256, 0, stream>>>(t0, qb, SD, (int)SDsz);
    gemm(context, cw_k + wOff, ctk, CTX, DD, DD, nullptr, 0);
    gemm(context, cw_v + wOff, ctv, CTX, DD, DD, nullptr, 0);
    {
      int tot = NH * CTX * HDIM;
      convk_kernel<<<(tot + 255) / 256, 256, 0, stream>>>(ctk, ckb, CTX, CTX, CTX, tot);
      convv_kernel<<<(tot + 255) / 256, 256, 0, stream>>>(ctv, cvb, CTX, CTX, CTX, tot);
    }
    flash_attn_kernel<<<dim3(SD / 128, NH), 256, 0, stream>>>(qb, ckb, cvb, t2,
                                                              SD, CTX, CTX, scale);
    gemm(t2, cw_o + wOff, t3, SD, DD, DD, nullptr, 0);
    residual_kernel<<<(int)(SDsz / 256), 256, 0, stream>>>(x, t3, modl, t_emb, -1,
                                                           (int)SDsz);

    // ---- audio attention (per frame, 5 kv tokens padded to 32) ----------
    adaln_kernel<<<SD, 256, 0, stream>>>(x, h, SD, modl, t_emb, -1, -1);
    adaln_kernel<<<NFR * AUD, 256, 0, stream>>>(audio, aln, NFR * AUD, modl, t_emb,
                                                -1, -1);
    gemm(h, aw_q + wOff, t0, SD, DD, DD, nullptr, 0);
    convq_kernel<<<(int)(SDsz / 256), 256, 0, stream>>>(t0, qb, SD / NFR, (int)SDsz);
    gemm(aln, aw_k + wOff, audk, NFR * AUD, DD, DD, nullptr, 0);
    gemm(aln, aw_v + wOff, audv, NFR * AUD, DD, DD, nullptr, 0);
    {
      int tot = NFR * NH * AKPAD * HDIM;
      convk_kernel<<<(tot + 255) / 256, 256, 0, stream>>>(audk, akb, AKPAD, AUD, AUD, tot);
      convv_kernel<<<(tot + 255) / 256, 256, 0, stream>>>(audv, avb, AKPAD, AUD, AUD, tot);
    }
    flash_attn_kernel<<<dim3((SD / NFR) / 128, NFR * NH), 256, 0, stream>>>(
        qb, akb, avb, t2, SD / NFR, AKPAD, AUD, scale);
    gemm(t2, aw_o + wOff, t3, SD, DD, DD, nullptr, 0);
    residual_kernel<<<(int)(SDsz / 256), 256, 0, stream>>>(x, t3, modl, t_emb, -1,
                                                           (int)SDsz);

    // ---- MLP ------------------------------------------------------------
    adaln_kernel<<<SD, 256, 0, stream>>>(x, h, SD, modl, t_emb, 4, 3);
    gemm(h, w1 + mOff, mid, SD, 4 * DD, DD, b1 + (size_t)l * 4 * DD, 1);
    gemm(mid, w2 + mOff, t3, SD, DD, 4 * DD, b2 + (size_t)l * DD, 0);
    residual_kernel<<<(int)(SDsz / 256), 256, 0, stream>>>(x, t3, modl, t_emb, 5,
                                                           (int)SDsz);
  }
  (void)in_sizes; (void)n_in; (void)out_size; (void)ws_size;
}
